// MPCNNLite_41832981463386
// MI455X (gfx1250) — matrix operand order repaired and checked by reading, not compiled, not run on hardware
//
#include <hip/hip_runtime.h>
#include <cstdint>
#include <cstddef>

// ---------------- problem constants ----------------
#define B_SZ   512
#define LSEQ   64
#define DDIM   300
#define DP     320        // D padded to mult of 32 for attention K
#define D2     600
#define NF     300
#define NFP    304        // filters padded to mult of 16
#define CPAD   608        // 600 channels padded to 19*32 per tap
#define HOUT   512
#define NFEAT  3321
#define KFC    3328       // 3321 padded to 104*32

typedef __attribute__((ext_vector_type(16))) __bf16 v16bf;
typedef __attribute__((ext_vector_type(8)))  __bf16 v8bf;
typedef __attribute__((ext_vector_type(8)))  float  v8f;

// ---------------- small helpers ----------------
static __device__ __forceinline__ unsigned short f2bf(float f) {
    unsigned u = __float_as_uint(f);
    u += 0x7FFFu + ((u >> 16) & 1u);          // round-to-nearest-even
    return (unsigned short)(u >> 16);
}
// order-preserving float->uint key (for atomicMax over floats)
static __device__ __forceinline__ unsigned enc_f(float x) {
    unsigned u = __float_as_uint(x);
    return (u & 0x80000000u) ? ~u : (u | 0x80000000u);
}
static __device__ __forceinline__ float dec_f(unsigned k) {
    unsigned u = (k & 0x80000000u) ? (k & 0x7FFFFFFFu) : ~k;
    return __uint_as_float(u);
}

// A fragment: 16x32 bf16, row-major M x K, tile origin given, row stride in elems.
// Layout (matches B): lanes0-15 (M=0..15): K0..K0+15 contiguous
//                     lanes16-31:          K0+16..K0+31 contiguous
static __device__ __forceinline__ v16bf load_a_frag(const unsigned short* A, int ldk) {
    int lane = threadIdx.x & 31;
    int half = lane >> 4;
    int m    = lane & 15;
    const unsigned short* p = A + (size_t)m * ldk + 16 * half;
    return *reinterpret_cast<const v16bf*>(p);
}
// B fragment: 32x16 bf16 but stored as Bt (N x K row-major). lane n = col,
// lanes0-15 hold K0..K0+15 contiguous, lanes16-31 hold K0+16..31 -> one 32B read.
static __device__ __forceinline__ v16bf load_b_frag(const unsigned short* Bt, int ldk) {
    int lane = threadIdx.x & 31;
    int half = lane >> 4;
    int n    = lane & 15;
    const unsigned short* p = Bt + (size_t)n * ldk + 16 * half;
    return *reinterpret_cast<const v16bf*>(p);
}
static __device__ __forceinline__ v8f wmma_bf16(v16bf a, v16bf b, v8f c) {
    return __builtin_amdgcn_wmma_f32_16x16x32_bf16(false, a, false, b, (short)0, c, false, false);
}
#define V8F_ZERO {0.f,0.f,0.f,0.f,0.f,0.f,0.f,0.f}

// ---------------- kernels ----------------
__global__ void k_zero_u32(unsigned* p, int n) {
    int i = blockIdx.x * blockDim.x + threadIdx.x;
    if (i < n) p[i] = 0u;
}

// gather embeddings -> bf16 [2][B][L][DP] (zero padded D..DP)
__global__ void k_gather_sh(const int* s1, const int* s2, const float* emb,
                            unsigned short* sh) {
    int idx = blockIdx.x * blockDim.x + threadIdx.x;
    int total = 2 * B_SZ * LSEQ * DP;
    if (idx >= total) return;
    int dp = idx % DP;
    int r  = idx / DP;
    int t  = r % LSEQ;  r /= LSEQ;
    int b  = r % B_SZ;
    int s  = r / B_SZ;
    int tok = (s ? s2 : s1)[b * LSEQ + t];
    float v = (dp < DDIM) ? emb[(size_t)tok * DDIM + dp] : 0.f;
    sh[idx] = f2bf(v);
}

// per-token L2 norms (f32, from raw embeddings) -> [2][B][L]
__global__ void k_norms(const int* s1, const int* s2, const float* emb, float* nrm) {
    int idx = blockIdx.x * blockDim.x + threadIdx.x;
    int total = 2 * B_SZ * LSEQ;
    if (idx >= total) return;
    int t = idx % LSEQ;
    int r = idx / LSEQ;
    int b = r % B_SZ;
    int s = r / B_SZ;
    int tok = (s ? s2 : s1)[b * LSEQ + t];
    const float* e = emb + (size_t)tok * DDIM;
    float acc = 0.f;
    for (int d = 0; d < DDIM; ++d) acc += e[d] * e[d];
    nrm[idx] = sqrtf(acc);
}

// attention GEMM per batch: 64x64, K=320 via WMMA; fold row/col sums of
// attn = dot/(n1*n2) into sums buffer [rowsum(B*L) | colsum(B*L)].
__global__ void k_attn(const unsigned short* sh, const float* nrm, float* sums) {
    int b    = blockIdx.x;
    int wave = threadIdx.x >> 5;
    int lane = threadIdx.x & 31;
    int half = lane >> 4;
    int nL   = lane & 15;
    const unsigned short* A  = sh + ((size_t)(0 * B_SZ + b) * LSEQ) * DP;
    const unsigned short* Bt = sh + ((size_t)(1 * B_SZ + b) * LSEQ) * DP;
    float* rowsum = sums + (size_t)b * LSEQ;
    float* colsum = sums + (size_t)(B_SZ + b) * LSEQ;
    int mtile = wave;                       // 4 waves cover 64 rows
    for (int ntile = 0; ntile < 4; ++ntile) {
        v8f acc = V8F_ZERO;
        for (int k0 = 0; k0 < DP; k0 += 32) {
            v16bf a = load_a_frag(A + (size_t)mtile * 16 * DP + k0, DP);
            v16bf bb = load_b_frag(Bt + (size_t)ntile * 16 * DP + k0, DP);
            acc = wmma_bf16(a, bb, acc);
        }
        int n = ntile * 16 + nL;
        float n2v = nrm[(size_t)(B_SZ + b) * LSEQ + n];
        float colpart = 0.f;
        for (int r = 0; r < 8; ++r) {
            int m = mtile * 16 + r + 8 * half;
            float av = acc[r] / (nrm[(size_t)b * LSEQ + m] * n2v);
            colpart += av;
            float rv = av;
            rv += __shfl_xor(rv, 1, 32);
            rv += __shfl_xor(rv, 2, 32);
            rv += __shfl_xor(rv, 4, 32);
            rv += __shfl_xor(rv, 8, 32);
            if (nL == 0) atomicAdd(&rowsum[m], rv);
        }
        atomicAdd(&colsum[n], colpart);
    }
}

// softmax over L=64 per (sent,batch): aw = softmax(sums)
__global__ void k_softmax(const float* sums, float* aw) {
    __shared__ float sm[LSEQ];
    int bid = blockIdx.x;                   // 0..2B-1
    int tid = threadIdx.x;                  // 64 threads
    float x = sums[(size_t)bid * LSEQ + tid];
    sm[tid] = x; __syncthreads();
    for (int st = 32; st > 0; st >>= 1) { if (tid < st) sm[tid] = fmaxf(sm[tid], sm[tid + st]); __syncthreads(); }
    float mx = sm[0]; __syncthreads();
    float e = expf(x - mx);
    sm[tid] = e; __syncthreads();
    for (int st = 32; st > 0; st >>= 1) { if (tid < st) sm[tid] += sm[tid + st]; __syncthreads(); }
    aw[(size_t)bid * LSEQ + tid] = e / sm[0];
}

// build a = [aw*s ; s] bf16 [2][B][600][L]; fused "inf" max via encoded atomicMax
__global__ void k_abuild(const int* s1, const int* s2, const float* emb,
                         const float* aw, unsigned short* abf, unsigned* infEnc) {
    int idx = blockIdx.x * blockDim.x + threadIdx.x;
    int total = 2 * B_SZ * D2 * LSEQ;
    if (idx >= total) return;
    int t = idx % LSEQ;
    int r = idx / LSEQ;
    int c = r % D2;  r /= D2;
    int b = r % B_SZ;
    int s = r / B_SZ;
    int tok = (s ? s2 : s1)[b * LSEQ + t];
    int d = (c < NF) ? c : (c - NF);
    float e = emb[(size_t)tok * DDIM + d];
    float v = (c < NF) ? aw[(size_t)(s * B_SZ + b) * LSEQ + t] * e : e;
    abf[idx] = f2bf(v);
    atomicMax(&infEnc[s * B_SZ + b], enc_f(v));
}

// reorder conv weights: wbf[f][kq*608 + c] = w[f][c][kq]  (bf16, zero padded)
__global__ void k_wprep(const float* cw, unsigned short* wbf, int ws) {
    int Kp = CPAD * ws;
    int idx = blockIdx.x * blockDim.x + threadIdx.x;
    int total = NFP * Kp;
    if (idx >= total) return;
    int col = idx % Kp;
    int f   = idx / Kp;
    int kq  = col / CPAD;
    int c   = col % CPAD;
    float v = (f < NF && c < D2) ? cw[((size_t)f * D2 + c) * ws + kq] : 0.f;
    wbf[idx] = f2bf(v);
}

// wide conv as WMMA GEMM + fused (bias, max over time) with tanh deferred.
// grid: (ceil(outLen/16), B), block 128 (4 waves). dyn LDS: 16*Kp*2 bytes.
__global__ void k_conv(const unsigned short* abf, const unsigned short* wbf,
                       const float* cb, unsigned* blkEnc,
                       int ws, int Kp, int outLen) {
    extern __shared__ unsigned short xs[];  // [16][Kp] im2col tile
    int b  = blockIdx.y;
    int t0 = blockIdx.x * 16;
    int tid = threadIdx.x;
    int total = 16 * Kp;
    int pad = ws - 1;
    for (int idx = tid; idx < total; idx += blockDim.x) {
        int n  = idx / Kp;
        int kk = idx - n * Kp;
        int kq = kk / CPAD;
        int c  = kk - kq * CPAD;
        int t  = t0 + n + kq - pad;
        unsigned short v = 0;
        if (c < D2 && t >= 0 && t < LSEQ)
            v = abf[((size_t)b * D2 + c) * LSEQ + t];
        xs[idx] = v;
    }
    __syncthreads();
    int wave = tid >> 5, lane = tid & 31, half = lane >> 4, nL = lane & 15;
    for (int mtile = wave; mtile < NFP / 16; mtile += 4) {
        v8f acc = V8F_ZERO;
        for (int k0 = 0; k0 < Kp; k0 += 32) {
            v16bf a = load_a_frag(wbf + (size_t)mtile * 16 * Kp + k0, Kp);
            v16bf bb = load_b_frag(xs + k0, Kp);
            acc = wmma_bf16(a, bb, acc);
        }
        int t = t0 + nL;
        bool vn = (t < outLen);
        for (int r = 0; r < 8; ++r) {
            int f = mtile * 16 + r + 8 * half;
            float v = vn ? (acc[r] + cb[f < NF ? f : NF - 1]) : -3.0e38f;
            v = fmaxf(v, __shfl_xor(v, 1, 32));
            v = fmaxf(v, __shfl_xor(v, 2, 32));
            v = fmaxf(v, __shfl_xor(v, 4, 32));
            v = fmaxf(v, __shfl_xor(v, 8, 32));
            if (nL == 0 && f < NF)
                atomicMax(&blkEnc[(size_t)b * NF + f], enc_f(v));
        }
    }
}

// decode encoded maxima: tanh for conv blocks, raw for inf block
__global__ void k_decode(const unsigned* blkEnc, const unsigned* infEnc,
                         float* blkF, float* infF) {
    int nblk = 6 * B_SZ * NF;
    int idx = blockIdx.x * blockDim.x + threadIdx.x;
    if (idx < nblk) {
        blkF[idx] = tanhf(dec_f(blkEnc[idx]));
    } else if (idx < nblk + 2 * B_SZ) {
        int j = idx - nblk;
        infF[j] = dec_f(infEnc[j]);
    }
}

static __device__ float block_sum(float v, float* red, int tid) {
    red[tid] = v; __syncthreads();
    for (int st = 128; st > 0; st >>= 1) { if (tid < st) red[tid] += red[tid + st]; __syncthreads(); }
    float r = red[0]; __syncthreads();
    return r;
}

// per-batch feature assembly -> bf16 [B][KFC] (3321 feats + zero pad)
__global__ void k_features(const float* blkF, const float* infF,
                           unsigned short* featbf) {
    __shared__ float red[256];
    __shared__ float nrm6[6];
    int b = blockIdx.x, tid = threadIdx.x;
    const float* x[2][3];
    for (int s = 0; s < 2; ++s)
        for (int w = 0; w < 3; ++w)
            x[s][w] = blkF + ((size_t)(s * 3 + w) * B_SZ + b) * NF;
    float i1 = infF[b], i2 = infF[B_SZ + b];
    unsigned short* fb = featbf + (size_t)b * KFC;

    // algo1: horizontal (per-filter 4-vectors)
    for (int f = tid; f < NF; f += 256) {
        float a0 = x[0][0][f], a1 = x[0][1][f], a2 = x[0][2][f], a3 = i1;
        float b0 = x[1][0][f], b1 = x[1][1][f], b2 = x[1][2][f], b3 = i2;
        float dot = a0*b0 + a1*b1 + a2*b2 + a3*b3;
        float na = fmaxf(sqrtf(a0*a0 + a1*a1 + a2*a2 + a3*a3), 1e-8f);
        float nb = fmaxf(sqrtf(b0*b0 + b1*b1 + b2*b2 + b3*b3), 1e-8f);
        fb[f] = f2bf(dot / (na * nb));
        float d0 = a0-b0+1e-6f, d1 = a1-b1+1e-6f, d2 = a2-b2+1e-6f, d3 = a3-b3+1e-6f;
        fb[NF + f] = f2bf(sqrtf(d0*d0 + d1*d1 + d2*d2 + d3*d3));
    }
    // norms over NF for the 6 blocks
    for (int q = 0; q < 6; ++q) {
        const float* p = x[q / 3][q % 3];
        float s = 0.f;
        for (int f = tid; f < NF; f += 256) s += p[f] * p[f];
        float tot = block_sum(s, red, tid);
        if (tid == 0) nrm6[q] = sqrtf(tot);
        __syncthreads();
    }
    // algo2: 9 finite pairs
    for (int p = 0; p < 9; ++p) {
        int wi = p / 3, wj = p % 3;
        const float* u = x[0][wi];
        const float* v = x[1][wj];
        float pd = 0.f, pq = 0.f;
        for (int f = tid; f < NF; f += 256) {
            float du = u[f], dv = v[f];
            pd += du * dv;
            float dd = du - dv + 1e-6f;
            pq += dd * dd;
        }
        float dot = block_sum(pd, red, tid);
        float dq  = block_sum(pq, red, tid);
        if (tid == 0) {
            float c = dot / (fmaxf(nrm6[wi], 1e-8f) * fmaxf(nrm6[3 + wj], 1e-8f));
            fb[2 * NF + p * (NF + 2) + 0] = f2bf(c);
            fb[2 * NF + p * (NF + 2) + 1] = f2bf(sqrtf(dq));
        }
        for (int f = tid; f < NF; f += 256)
            fb[2 * NF + p * (NF + 2) + 2 + f] = f2bf(fabsf(u[f] - v[f]));
        __syncthreads();
    }
    // inf-inf pair
    if (tid == 0) {
        float ni = fmaxf(fabsf(i1), 1e-8f), nj = fmaxf(fabsf(i2), 1e-8f);
        fb[3318] = f2bf((i1 * i2) / (ni * nj));
        float dd = i1 - i2 + 1e-6f;
        fb[3319] = f2bf(sqrtf(dd * dd));
        fb[3320] = f2bf(fabsf(i1 - i2));
    }
    for (int f = NFEAT + tid; f < KFC; f += 256) fb[f] = 0;
}

// fc_w (3321 x 512) -> transposed bf16 (512 x KFC), zero padded K
__global__ void k_fcprep(const float* fc_w, unsigned short* fcwT) {
    int idx = blockIdx.x * blockDim.x + threadIdx.x;
    int total = HOUT * KFC;
    if (idx >= total) return;
    int k = idx % KFC;
    int n = idx / KFC;
    float v = (k < NFEAT) ? fc_w[(size_t)k * HOUT + n] : 0.f;
    fcwT[idx] = f2bf(v);
}

// out = tanh(feat @ fc_w + fc_b) via WMMA. grid (HOUT/16, B/16), 1 wave.
__global__ void k_fc(const unsigned short* featbf, const unsigned short* fcwT,
                     const float* fc_b, float* out) {
    int n0 = blockIdx.x * 16;
    int m0 = blockIdx.y * 16;
    int lane = threadIdx.x & 31;
    int half = lane >> 4;
    int nL = lane & 15;
    v8f acc = V8F_ZERO;
    for (int k0 = 0; k0 < KFC; k0 += 32) {
        v16bf a = load_a_frag(featbf + (size_t)m0 * KFC + k0, KFC);
        v16bf bb = load_b_frag(fcwT + (size_t)n0 * KFC + k0, KFC);
        acc = wmma_bf16(a, bb, acc);
    }
    int n = n0 + nL;
    float bias = fc_b[n];
    for (int r = 0; r < 8; ++r) {
        int m = m0 + r + 8 * half;
        out[(size_t)m * HOUT + n] = tanhf(acc[r] + bias);
    }
}

// ---------------- host launcher ----------------
extern "C" void kernel_launch(void* const* d_in, const int* in_sizes, int n_in,
                              void* d_out, int out_size, void* d_ws, size_t ws_size,
                              hipStream_t stream) {
    (void)in_sizes; (void)n_in; (void)out_size; (void)ws_size;
    const int*   sent1 = (const int*)d_in[0];
    const int*   sent2 = (const int*)d_in[1];
    const float* emb   = (const float*)d_in[2];
    const float* fc_w  = (const float*)d_in[3];
    const float* fc_b  = (const float*)d_in[4];
    const float* cw[3] = { (const float*)d_in[5], (const float*)d_in[7], (const float*)d_in[9] };
    const float* cb[3] = { (const float*)d_in[6], (const float*)d_in[8], (const float*)d_in[10] };
    float* out = (float*)d_out;

    unsigned char* W = (unsigned char*)d_ws;
    size_t o = 0;
    auto take = [&](size_t bytes) -> void* {
        o = (o + 255) & ~(size_t)255;
        void* p = W + o;
        o += bytes;
        return p;
    };
    unsigned short* sh     = (unsigned short*)take((size_t)2 * B_SZ * LSEQ * DP * 2);
    float*          nrm    = (float*)take((size_t)2 * B_SZ * LSEQ * 4);
    float*          sums   = (float*)take((size_t)2 * B_SZ * LSEQ * 4);
    float*          aw     = (float*)take((size_t)2 * B_SZ * LSEQ * 4);
    unsigned short* abf    = (unsigned short*)take((size_t)2 * B_SZ * D2 * LSEQ * 2);
    unsigned short* wbf[3];
    wbf[0] = (unsigned short*)take((size_t)NFP * (CPAD * 1) * 2);
    wbf[1] = (unsigned short*)take((size_t)NFP * (CPAD * 2) * 2);
    wbf[2] = (unsigned short*)take((size_t)NFP * (CPAD * 3) * 2);
    unsigned*       blkEnc = (unsigned*)take((size_t)6 * B_SZ * NF * 4);
    unsigned*       infEnc = (unsigned*)take((size_t)2 * B_SZ * 4);
    float*          blkF   = (float*)take((size_t)6 * B_SZ * NF * 4);
    float*          infF   = (float*)take((size_t)2 * B_SZ * 4);
    unsigned short* featbf = (unsigned short*)take((size_t)B_SZ * KFC * 2);
    unsigned short* fcwT   = (unsigned short*)take((size_t)HOUT * KFC * 2);

    // zero the accumulation buffers (sums, blkEnc, infEnc)
    {
        int n1 = 2 * B_SZ * LSEQ;
        k_zero_u32<<<(n1 + 255) / 256, 256, 0, stream>>>((unsigned*)sums, n1);
        int n2 = 6 * B_SZ * NF;
        k_zero_u32<<<(n2 + 255) / 256, 256, 0, stream>>>(blkEnc, n2);
        int n3 = 2 * B_SZ;
        k_zero_u32<<<(n3 + 255) / 256, 256, 0, stream>>>(infEnc, n3);
    }
    // gather + norms
    {
        int tot = 2 * B_SZ * LSEQ * DP;
        k_gather_sh<<<(tot + 255) / 256, 256, 0, stream>>>(sent1, sent2, emb, sh);
        int tn = 2 * B_SZ * LSEQ;
        k_norms<<<(tn + 255) / 256, 256, 0, stream>>>(sent1, sent2, emb, nrm);
    }
    // attention + softmax
    k_attn<<<B_SZ, 128, 0, stream>>>(sh, nrm, sums);
    k_softmax<<<2 * B_SZ, LSEQ, 0, stream>>>(sums, aw);
    // build a-matrices (+ inf max)
    {
        int tot = 2 * B_SZ * D2 * LSEQ;
        k_abuild<<<(tot + 255) / 256, 256, 0, stream>>>(sent1, sent2, emb, aw, abf, infEnc);
    }
    // conv weight prep
    for (int w = 0; w < 3; ++w) {
        int ws = w + 1;
        int tot = NFP * CPAD * ws;
        k_wprep<<<(tot + 255) / 256, 256, 0, stream>>>(cw[w], wbf[w], ws);
    }
    // convolutions: 3 widths x 2 sentences
    for (int w = 0; w < 3; ++w) {
        int ws = w + 1;
        int Kp = CPAD * ws;
        int outLen = LSEQ + ws - 1;
        int ntiles = (outLen + 15) / 16;
        size_t shb = (size_t)16 * Kp * 2;
        for (int s = 0; s < 2; ++s) {
            const unsigned short* a_s = abf + (size_t)s * B_SZ * D2 * LSEQ;
            unsigned* enc_sw = blkEnc + (size_t)(s * 3 + w) * B_SZ * NF;
            dim3 grid(ntiles, B_SZ);
            k_conv<<<grid, 128, shb, stream>>>(a_s, wbf[w], cb[w], enc_sw, ws, Kp, outLen);
        }
    }
    // decode maxima (tanh for conv blocks)
    {
        int tot = 6 * B_SZ * NF + 2 * B_SZ;
        k_decode<<<(tot + 255) / 256, 256, 0, stream>>>(blkEnc, infEnc, blkF, infF);
    }
    // features + FC
    k_features<<<B_SZ, 256, 0, stream>>>(blkF, infF, featbf);
    {
        int tot = HOUT * KFC;
        k_fcprep<<<(tot + 255) / 256, 256, 0, stream>>>(fc_w, fcwT);
    }
    {
        dim3 grid(HOUT / 16, B_SZ / 16);
        k_fc<<<grid, 32, 0, stream>>>(featbf, fcwT, fc_b, out);
    }
}
